// ProjectRiskGNN_31842887533309
// MI455X (gfx1250) — matrix operand order repaired and checked by reading, not compiled, not run on hardware
//
#include <hip/hip_runtime.h>
#include <hip/hip_bf16.h>

typedef __attribute__((ext_vector_type(16))) _Float16 v16h;
typedef __attribute__((ext_vector_type(8)))  _Float16 v8h;
typedef __attribute__((ext_vector_type(8)))  float    v8f;

#define N_NODES  100000
#define N_EDGES  1600000
#define N_GRAPHS 64
#define IN_CH    128
#define HID      64
#define ROWTILES (N_NODES / 16)   // 6250

// ---------------------------------------------------------------------------
// Zero a float buffer
// ---------------------------------------------------------------------------
__global__ void gnn_zero_f32(float* __restrict__ p, int n) {
  int i = blockIdx.x * blockDim.x + threadIdx.x;
  if (i < n) p[i] = 0.0f;
}

// ---------------------------------------------------------------------------
// Pack a [K x 64] f32 weight matrix into the exact WMMA B-operand register
// layout as f16, so the GEMM loads one aligned v16h per (colTile, kChunk, lane):
//   packed[(((ct*(K/32) + c)*32 + lane)*16 + t]
//     = W[(c*32 + (t>>3)*16 + (lane>>4)*8 + (t&7)) * 64 + ct*16 + (lane&15)]
// ---------------------------------------------------------------------------
__global__ void gnn_pack_w(const float* __restrict__ w,
                           _Float16* __restrict__ packed, int K) {
  int idx = blockIdx.x * blockDim.x + threadIdx.x;
  if (idx >= K * HID) return;
  int t    = idx & 15;
  int l    = (idx >> 4) & 31;
  int rest = idx >> 9;
  int nchunk = K >> 5;
  int c  = rest % nchunk;
  int ct = rest / nchunk;
  int k  = c * 32 + ((t >> 3) << 4) + ((l >> 4) << 3) + (t & 7);
  int n  = ct * 16 + (l & 15);
  packed[idx] = (_Float16)w[(size_t)k * HID + n];
}

// ---------------------------------------------------------------------------
// Layer-0 dual GEMM (f32 input, K = 128):
//   P(f16) = IN @ Wrel,  R(f32) = IN @ Wroot
// One wave owns a 16-row x 64-col output strip (4 col tiles, 8 accumulators);
// block = 8 waves = 8 row tiles. A loaded once per K-chunk, B tiles are
// pre-packed v16h loads. K-chunk loop fully unrolled.
// ---------------------------------------------------------------------------
__global__ __launch_bounds__(256)
void gnn_gemm_dual_l0(const float* __restrict__ in,
                      const _Float16* __restrict__ wrelP,
                      const _Float16* __restrict__ wrootP,
                      _Float16* __restrict__ P,
                      float* __restrict__ R) {
  constexpr int K = IN_CH;
  constexpr int NCH = K / 32;
  const int lane    = threadIdx.x & 31;
  const int wv      = threadIdx.x >> 5;
  const int rowTile = blockIdx.x * 8 + wv;
  if (rowTile >= ROWTILES) return;
  const int rowBase = rowTile * 16;
  const int m = lane & 15;
  const int g = lane >> 4;

  v8f accP[4] = {};
  v8f accR[4] = {};

  const float* rowPtr = in + (size_t)(rowBase + m) * K;

#pragma unroll
  for (int c = 0; c < NCH; ++c) {
    // A tile 16x32 f16: lanes 0-15 halfs = K{0-7,16-23}; lanes 16-31 = K{8-15,24-31}
    v16h a;
    {
      const float4 lo0 = *(const float4*)(rowPtr + c * 32 + g * 8);
      const float4 lo1 = *(const float4*)(rowPtr + c * 32 + g * 8 + 4);
      const float4 hi0 = *(const float4*)(rowPtr + c * 32 + 16 + g * 8);
      const float4 hi1 = *(const float4*)(rowPtr + c * 32 + 16 + g * 8 + 4);
      a[0]  = (_Float16)lo0.x; a[1]  = (_Float16)lo0.y;
      a[2]  = (_Float16)lo0.z; a[3]  = (_Float16)lo0.w;
      a[4]  = (_Float16)lo1.x; a[5]  = (_Float16)lo1.y;
      a[6]  = (_Float16)lo1.z; a[7]  = (_Float16)lo1.w;
      a[8]  = (_Float16)hi0.x; a[9]  = (_Float16)hi0.y;
      a[10] = (_Float16)hi0.z; a[11] = (_Float16)hi0.w;
      a[12] = (_Float16)hi1.x; a[13] = (_Float16)hi1.y;
      a[14] = (_Float16)hi1.z; a[15] = (_Float16)hi1.w;
    }
#pragma unroll
    for (int ct = 0; ct < 4; ++ct) {
      const size_t boff = ((size_t)(ct * NCH + c) * 32 + lane) * 16;
      const v16h brel  = *(const v16h*)(wrelP  + boff);
      const v16h broot = *(const v16h*)(wrootP + boff);
      accP[ct] = __builtin_amdgcn_wmma_f32_16x16x32_f16(
          false, a, false, brel, (short)0, accP[ct], false, false);
      accR[ct] = __builtin_amdgcn_wmma_f32_16x16x32_f16(
          false, a, false, broot, (short)0, accR[ct], false, false);
    }
  }

#pragma unroll
  for (int ct = 0; ct < 4; ++ct) {
#pragma unroll
    for (int r = 0; r < 8; ++r) {
      const size_t off = (size_t)(rowBase + g * 8 + r) * HID + ct * 16 + m;
      P[off] = (_Float16)accP[ct][r];
      R[off] = accR[ct][r];
    }
  }
}

// ---------------------------------------------------------------------------
// Hidden-layer dual GEMM (f16 input, K = 64): same structure, A loads are
// direct v8h (b128) loads, zero converts in the hot loop.
// ---------------------------------------------------------------------------
__global__ __launch_bounds__(256)
void gnn_gemm_dual_f16(const _Float16* __restrict__ in,
                       const _Float16* __restrict__ wrelP,
                       const _Float16* __restrict__ wrootP,
                       _Float16* __restrict__ P,
                       float* __restrict__ R) {
  constexpr int K = HID;
  constexpr int NCH = K / 32;
  const int lane    = threadIdx.x & 31;
  const int wv      = threadIdx.x >> 5;
  const int rowTile = blockIdx.x * 8 + wv;
  if (rowTile >= ROWTILES) return;
  const int rowBase = rowTile * 16;
  const int m = lane & 15;
  const int g = lane >> 4;

  v8f accP[4] = {};
  v8f accR[4] = {};

  const _Float16* rowPtr = in + (size_t)(rowBase + m) * K;

#pragma unroll
  for (int c = 0; c < NCH; ++c) {
    v16h a;
    {
      const v8h lo = *(const v8h*)(rowPtr + c * 32 + g * 8);
      const v8h hi = *(const v8h*)(rowPtr + c * 32 + 16 + g * 8);
#pragma unroll
      for (int t = 0; t < 8; ++t) { a[t] = lo[t]; a[8 + t] = hi[t]; }
    }
#pragma unroll
    for (int ct = 0; ct < 4; ++ct) {
      const size_t boff = ((size_t)(ct * NCH + c) * 32 + lane) * 16;
      const v16h brel  = *(const v16h*)(wrelP  + boff);
      const v16h broot = *(const v16h*)(wrootP + boff);
      accP[ct] = __builtin_amdgcn_wmma_f32_16x16x32_f16(
          false, a, false, brel, (short)0, accP[ct], false, false);
      accR[ct] = __builtin_amdgcn_wmma_f32_16x16x32_f16(
          false, a, false, broot, (short)0, accR[ct], false, false);
    }
  }

#pragma unroll
  for (int ct = 0; ct < 4; ++ct) {
#pragma unroll
    for (int r = 0; r < 8; ++r) {
      const size_t off = (size_t)(rowBase + g * 8 + r) * HID + ct * 16 + m;
      P[off] = (_Float16)accP[ct][r];
      R[off] = accR[ct][r];
    }
  }
}

// ---------------------------------------------------------------------------
// Edge scatter-add: A[dst] += P[src]  (P is f16, packed b128 gathers;
// accumulation in f32 atomics). 4 threads per edge, 16 channels each.
// ---------------------------------------------------------------------------
__global__ __launch_bounds__(256)
void gnn_edge_scatter(const int* __restrict__ src,
                      const int* __restrict__ dst,
                      const _Float16* __restrict__ P,
                      float* __restrict__ A,
                      int nEdges) {
  int gid  = blockIdx.x * blockDim.x + threadIdx.x;
  int e    = gid >> 2;
  int part = gid & 3;
  if (e >= nEdges) return;
  int s = src[e];
  int d = dst[e];
  const v8h* pv = (const v8h*)(P + (size_t)s * HID + part * 16);
  float*     av = A + (size_t)d * HID + part * 16;
  const v8h v0 = pv[0];
  const v8h v1 = pv[1];
#pragma unroll
  for (int j = 0; j < 8; ++j) {
    atomicAdd(av + j,     (float)v0[j]);
    atomicAdd(av + 8 + j, (float)v1[j]);
  }
}

// ---------------------------------------------------------------------------
// h(f16) = relu(agg + bias + root)
// ---------------------------------------------------------------------------
__global__ __launch_bounds__(256)
void gnn_combine_relu(const float* __restrict__ agg,
                      const float* __restrict__ root,
                      const float* __restrict__ bias,
                      _Float16* __restrict__ h,
                      int n) {
  int i = blockIdx.x * blockDim.x + threadIdx.x;
  if (i >= n) return;
  int c = i & (HID - 1);
  float v = agg[i] + bias[c] + root[i];
  h[i] = (_Float16)(v > 0.0f ? v : 0.0f);
}

// ---------------------------------------------------------------------------
// Mean-pool accumulation (f32 atomics)
// ---------------------------------------------------------------------------
__global__ __launch_bounds__(256)
void gnn_pool_accum(const _Float16* __restrict__ h,
                    const int* __restrict__ batch,
                    float* __restrict__ sums,
                    float* __restrict__ cnts,
                    int n) {
  int i = blockIdx.x * blockDim.x + threadIdx.x;
  if (i >= n) return;
  int node = i >> 6;
  int c    = i & (HID - 1);
  int grp  = batch[node];
  atomicAdd(&sums[(size_t)grp * HID + c], (float)h[i]);
  if (c == 0) atomicAdd(&cnts[grp], 1.0f);
}

// ---------------------------------------------------------------------------
// MLP head: one block per graph, 64 threads
// ---------------------------------------------------------------------------
__global__ __launch_bounds__(64)
void gnn_head(const float* __restrict__ sums,
              const float* __restrict__ cnts,
              const float* __restrict__ w1, const float* __restrict__ b1,
              const float* __restrict__ w2, const float* __restrict__ b2,
              float* __restrict__ out) {
  __shared__ float pooled[HID];
  __shared__ float z[HID];
  const int grp = blockIdx.x;
  const int t   = threadIdx.x;

  float c = cnts[grp];
  c = c > 1.0f ? c : 1.0f;
  pooled[t] = sums[(size_t)grp * HID + t] / c;
  __syncthreads();

  float acc = b1[t];
#pragma unroll 8
  for (int k = 0; k < HID; ++k) acc += pooled[k] * w1[k * HID + t];
  z[t] = acc > 0.0f ? acc : 0.0f;
  __syncthreads();

  if (t < 2) {
    float o = b2[t];
#pragma unroll 8
    for (int k = 0; k < HID; ++k) o += z[k] * w2[k * 2 + t];
    out[grp * 2 + t] = o;
  }
}

// ---------------------------------------------------------------------------
// Launch
// ---------------------------------------------------------------------------
extern "C" void kernel_launch(void* const* d_in, const int* in_sizes, int n_in,
                              void* d_out, int out_size, void* d_ws, size_t ws_size,
                              hipStream_t stream) {
  const float* x        = (const float*)d_in[0];
  const int*   edge_idx = (const int*)d_in[1];
  const int*   batch    = (const int*)d_in[2];
  const float* w_rel0   = (const float*)d_in[3];
  const float* b_rel0   = (const float*)d_in[4];
  const float* w_root0  = (const float*)d_in[5];
  const float* w_rel1   = (const float*)d_in[6];
  const float* b_rel1   = (const float*)d_in[7];
  const float* w_root1  = (const float*)d_in[8];
  const float* w_rel2   = (const float*)d_in[9];
  const float* b_rel2   = (const float*)d_in[10];
  const float* w_root2  = (const float*)d_in[11];
  const float* head_w1  = (const float*)d_in[12];
  const float* head_b1  = (const float*)d_in[13];
  const float* head_w2  = (const float*)d_in[14];
  const float* head_b2  = (const float*)d_in[15];
  float* out = (float*)d_out;

  const int* e_src = edge_idx;            // row 0
  const int* e_dst = edge_idx + N_EDGES;  // row 1

  // ---- workspace layout (f32 buffers first, then f16) ----
  const size_t S = (size_t)N_NODES * HID;
  float* R    = (float*)d_ws;                       // 25.6 MB
  float* Agg  = R + S;                              // 25.6 MB
  float* sums = Agg + S;                            // 16 KB
  float* cnts = sums + (size_t)N_GRAPHS * HID;      // 256 B
  _Float16* P16 = (_Float16*)(cnts + N_GRAPHS);     // 12.8 MB
  _Float16* H16 = P16 + S;                          // 12.8 MB
  _Float16* pRel0  = H16 + S;                       // packed weights (f16)
  _Float16* pRoot0 = pRel0  + (size_t)IN_CH * HID;
  _Float16* pRel1  = pRoot0 + (size_t)IN_CH * HID;
  _Float16* pRoot1 = pRel1  + (size_t)HID * HID;
  _Float16* pRel2  = pRoot1 + (size_t)HID * HID;
  _Float16* pRoot2 = pRel2  + (size_t)HID * HID;

  const int NH       = N_NODES * HID;
  const int zeroGrid = (NH + 255) / 256;
  const int scatGrid = (N_EDGES * 4 + 255) / 256;
  const int gemmGrid = (ROWTILES + 7) / 8;          // 782

  // ---- pack weights into WMMA B-operand layout (tiny) ----
  gnn_pack_w<<<(IN_CH * HID + 255) / 256, 256, 0, stream>>>(w_rel0,  pRel0,  IN_CH);
  gnn_pack_w<<<(IN_CH * HID + 255) / 256, 256, 0, stream>>>(w_root0, pRoot0, IN_CH);
  gnn_pack_w<<<(HID   * HID + 255) / 256, 256, 0, stream>>>(w_rel1,  pRel1,  HID);
  gnn_pack_w<<<(HID   * HID + 255) / 256, 256, 0, stream>>>(w_root1, pRoot1, HID);
  gnn_pack_w<<<(HID   * HID + 255) / 256, 256, 0, stream>>>(w_rel2,  pRel2,  HID);
  gnn_pack_w<<<(HID   * HID + 255) / 256, 256, 0, stream>>>(w_root2, pRoot2, HID);

  // ---- Layer 0 (K = 128, f32 input) ----
  gnn_gemm_dual_l0<<<gemmGrid, 256, 0, stream>>>(x, pRel0, pRoot0, P16, R);
  gnn_zero_f32<<<zeroGrid, 256, 0, stream>>>(Agg, NH);
  gnn_edge_scatter<<<scatGrid, 256, 0, stream>>>(e_src, e_dst, P16, Agg, N_EDGES);
  gnn_combine_relu<<<zeroGrid, 256, 0, stream>>>(Agg, R, b_rel0, H16, NH);

  // ---- Layer 1 (K = 64, f16 input) ----
  gnn_gemm_dual_f16<<<gemmGrid, 256, 0, stream>>>(H16, pRel1, pRoot1, P16, R);
  gnn_zero_f32<<<zeroGrid, 256, 0, stream>>>(Agg, NH);
  gnn_edge_scatter<<<scatGrid, 256, 0, stream>>>(e_src, e_dst, P16, Agg, N_EDGES);
  gnn_combine_relu<<<zeroGrid, 256, 0, stream>>>(Agg, R, b_rel1, H16, NH);

  // ---- Layer 2 (K = 64, f16 input) ----
  gnn_gemm_dual_f16<<<gemmGrid, 256, 0, stream>>>(H16, pRel2, pRoot2, P16, R);
  gnn_zero_f32<<<zeroGrid, 256, 0, stream>>>(Agg, NH);
  gnn_edge_scatter<<<scatGrid, 256, 0, stream>>>(e_src, e_dst, P16, Agg, N_EDGES);
  gnn_combine_relu<<<zeroGrid, 256, 0, stream>>>(Agg, R, b_rel2, H16, NH);

  // ---- Mean pool + head ----
  gnn_zero_f32<<<(N_GRAPHS * HID + N_GRAPHS + 255) / 256, 256, 0, stream>>>(
      sums, N_GRAPHS * HID + N_GRAPHS);
  gnn_pool_accum<<<zeroGrid, 256, 0, stream>>>(H16, batch, sums, cnts, NH);
  gnn_head<<<N_GRAPHS, 64, 0, stream>>>(sums, cnts, head_w1, head_b1,
                                        head_w2, head_b2, out);
}
